// MSRSA_60232621359304
// MI455X (gfx1250) — matrix operand order repaired and checked by reading, not compiled, not run on hardware
//
#include <hip/hip_runtime.h>
#include <hip/hip_bf16.h>

typedef __attribute__((ext_vector_type(16))) _Float16 v16h;
typedef __attribute__((ext_vector_type(8)))  float    v8f;
typedef __attribute__((ext_vector_type(4)))  float    v4f;

#define Bb   8
#define Hh   8
#define Nn   1024
#define Dd   64
#define CH   128            // K/V chunk rows staged in LDS
#define NCH  (Nn / CH)      // 8 chunks

// D = A(16x32 f16) * B(32x16 f16) + C(16x16 f32)
#define WMMA_F16(A, B, C) \
  __builtin_amdgcn_wmma_f32_16x16x32_f16(false, (A), false, (B), (short)0, (C), false, false)

__global__ __launch_bounds__(256)
void msrsa_attn_kernel(const float* __restrict__ Q,
                       const float* __restrict__ K,
                       const float* __restrict__ V,
                       const unsigned char* __restrict__ mask,  // bool (B,1,N,N)
                       const float* __restrict__ adj,           // (B,N,N)
                       const float* __restrict__ wA,            // (1,H,1,1)
                       float* __restrict__ out,                 // (B,H,N,D)
                       float* __restrict__ attn) {              // (B,H,N,N)
  // ---- LDS (~49 KB total, under 64 KB) ----
  __shared__ _Float16 Kst[CH][Dd];     // 16 KB : K chunk, f16, row-major
  __shared__ _Float16 Vt [Dd][CH];     // 16 KB : V chunk, f16, transposed [d][krow]
  __shared__ float    Pst[16][CH];     //  8 KB : P chunk f32 (reused as out-partials)
  __shared__ float    Fst[16][CH];     //  8 KB : combined mask/adjacency/scale factor
  __shared__ float    wsums[8][16];
  __shared__ float    rowrcp[16];

  const int tid  = threadIdx.x;
  const int w    = tid >> 5;      // wave id 0..7 (wave32)
  const int lane = tid & 31;
  const int lh   = lane >> 4;     // half-wave group (C/D layout: row = r + 8*lh)
  const int l15  = lane & 15;

  const int bid  = blockIdx.x;    // 64 (b,h) * 64 row-tiles
  const int bh   = bid >> 6;
  const int rt   = bid & 63;
  const int b    = bh >> 3;
  const int h    = bh & 7;
  const int row0 = rt * 16;

  const float* Qp = Q + ((size_t)bh * Nn + row0) * Dd;
  const float* Kp = K + (size_t)bh * Nn * Dd;
  const float* Vp = V + (size_t)bh * Nn * Dd;
  const unsigned char* Mp = mask + (size_t)b * Nn * Nn + (size_t)row0 * Nn;
  const float* Ap = adj  + (size_t)b * Nn * Nn + (size_t)row0 * Nn;
  float* outp  = out  + ((size_t)bh * Nn + row0) * Dd;
  float* attnp = attn + (size_t)bh * Nn * Nn + (size_t)row0 * Nn;
  const float wa = wA[h];

  // ---- Q A-fragments (16x32 f16 x2, covering D=64), in contiguous k order:
// lane: M = lane&15 ; element e: k = 16*lh + e (contiguous, matches B fragments)
  v16h qa0, qa1;
  #pragma unroll
  for (int e = 0; e < 16; ++e) {
    const int v = e >> 1, p = e & 1;
    const int base = lh * 8;
    const int d0 = 2 * base + e;
    qa0[e] = (_Float16)Qp[l15 * Dd + d0];
    qa1[e] = (_Float16)Qp[l15 * Dd + d0 + 32];
  }

  // ================= PASS 1: row sums of exp(s)  (softmax without max-shift;
  // scores are bounded well below f32-exp overflow, clamp guards the tail) ======
  float ol[8];
  #pragma unroll
  for (int r = 0; r < 8; ++r) ol[r] = 0.f;

  for (int c = 0; c < NCH; ++c) {
    const int kbase = c * CH;
    // stage K chunk (coalesced) ...
    for (int i = tid; i < CH * Dd; i += 256)
      Kst[i >> 6][i & 63] = (_Float16)Kp[(size_t)(kbase + (i >> 6)) * Dd + (i & 63)];
    // ... and branchless mask/adjacency factor tile (coalesced u8 + f32 loads)
    for (int i = tid; i < 16 * CH; i += 256) {
      const int rr = i >> 7, cc = i & 127;
      const float mv = (float)Mp[(size_t)rr * Nn + kbase + cc];
      const float av = Ap[(size_t)rr * Nn + kbase + cc];
      Fst[rr][cc] = mv * (1.f + av * wa) * 0.125f;   // 0 when masked; includes 1/sqrt(D)
    }
    if (c + 1 < NCH)
      __builtin_prefetch(Kp + (size_t)(kbase + CH) * Dd + tid * 8, 0, 1);
    __syncthreads();

    // B-fragments of K^T: lane -> col n = l15 ; halves -> k(d) = 16*lh + h (+32 for frag 1)
    v16h fb0 = *(const v16h*)&Kst[w * 16 + l15][lh * 16];
    v16h fb1 = *(const v16h*)&Kst[w * 16 + l15][lh * 16 + 32];
    v8f s = {};
    s = WMMA_F16(qa0, fb0, s);
    s = WMMA_F16(qa1, fb1, s);

    #pragma unroll
    for (int r = 0; r < 8; ++r) {
      const int mrow = r + 8 * lh;
      const float sv = s[r] * Fst[mrow][w * 16 + l15];
      ol[r] += __expf(fminf(sv, 80.f));
    }
    __syncthreads();
  }

  // sum across the 16 lanes of each half-wave (xor steps stay within the group)
  #pragma unroll
  for (int st = 1; st < 16; st <<= 1) {
    #pragma unroll
    for (int r = 0; r < 8; ++r) ol[r] += __shfl_xor(ol[r], st, 32);
  }
  if (l15 == 0) {
    #pragma unroll
    for (int r = 0; r < 8; ++r) wsums[w][r + 8 * lh] = ol[r];
  }
  __syncthreads();
  if (tid < 16) {
    float L0 = 0.f;
    #pragma unroll
    for (int ww = 0; ww < 8; ++ww) L0 += wsums[ww][tid];
    rowrcp[tid] = 1.f / L0;
  }
  __syncthreads();

  // ============== PASS 2: recompute S, emit attn_weight, fuse P @ V ==============
  v8f o = {};
  const int ntile = w & 3;   // which 16-wide slice of D this wave produces
  const int khalf = w >> 2;  // which 64-row half of each chunk this wave reduces

  for (int c = 0; c < NCH; ++c) {
    const int kbase = c * CH;
    for (int i = tid; i < CH * Dd; i += 256) {
      const int r = i >> 6, dc = i & 63;
      Kst[r][dc] = (_Float16)Kp[(size_t)(kbase + r) * Dd + dc];
      Vt[dc][r]  = (_Float16)Vp[(size_t)(kbase + r) * Dd + dc];  // transpose into LDS
    }
    for (int i = tid; i < 16 * CH; i += 256) {
      const int rr = i >> 7, cc = i & 127;
      const float mv = (float)Mp[(size_t)rr * Nn + kbase + cc];
      const float av = Ap[(size_t)rr * Nn + kbase + cc];
      Fst[rr][cc] = mv * (1.f + av * wa) * 0.125f;
    }
    if (c + 1 < NCH) {
      __builtin_prefetch(Kp + (size_t)(kbase + CH) * Dd + tid * 8, 0, 1);
      __builtin_prefetch(Vp + (size_t)(kbase + CH) * Dd + tid * 8, 0, 1);
    }
    __syncthreads();

    v16h fb0 = *(const v16h*)&Kst[w * 16 + l15][lh * 16];
    v16h fb1 = *(const v16h*)&Kst[w * 16 + l15][lh * 16 + 32];
    v8f s = {};
    s = WMMA_F16(qa0, fb0, s);
    s = WMMA_F16(qa1, fb1, s);

    #pragma unroll
    for (int r = 0; r < 8; ++r) {
      const int mrow = r + 8 * lh;
      const float sv = s[r] * Fst[mrow][w * 16 + l15];
      Pst[mrow][w * 16 + l15] = __expf(fminf(sv, 80.f)) * rowrcp[mrow];
    }
    __syncthreads();

    // streaming (non-temporal) float4 store of the 16x128 attn_weight chunk
    for (int i = tid; i < 16 * (CH / 4); i += 256) {
      const int rr = i >> 5;
      const int c4 = (i & 31) * 4;
      __builtin_nontemporal_store(*(const v4f*)&Pst[rr][c4],
                                  (v4f*)&attnp[(size_t)rr * Nn + kbase + c4]);
    }

    // P @ V : this wave reduces 64 K-rows of the chunk into its 16-wide D slice
    #pragma unroll
    for (int s2 = 0; s2 < 2; ++s2) {
      const int k0 = khalf * 64 + s2 * 32;
      v16h pa;
      #pragma unroll
      for (int e = 0; e < 16; ++e) {
        const int v = e >> 1, p = e & 1;
        const int base = lh * 8;
        const int koff = 2 * base + e;
        pa[e] = (_Float16)Pst[l15][k0 + koff];
      }
      v16h vb = *(const v16h*)&Vt[ntile * 16 + l15][k0 + lh * 16];
      o = WMMA_F16(pa, vb, o);
    }
    __syncthreads();
  }

  // combine the two K-half partials through LDS (reuse Pst: 2 * 16 * 64 floats)
  float* sp = &Pst[0][0];
  #pragma unroll
  for (int r = 0; r < 8; ++r)
    sp[khalf * 1024 + (r + 8 * lh) * 64 + ntile * 16 + l15] = o[r];
  __syncthreads();
  for (int i = tid; i < 16 * Dd; i += 256)
    outp[i] = sp[i] + sp[1024 + i];
}

extern "C" void kernel_launch(void* const* d_in, const int* in_sizes, int n_in,
                              void* d_out, int out_size, void* d_ws, size_t ws_size,
                              hipStream_t stream) {
  const float*         Qd   = (const float*)d_in[0];
  const float*         Kd   = (const float*)d_in[1];
  const float*         Vd   = (const float*)d_in[2];
  const unsigned char* Md   = (const unsigned char*)d_in[3];  // bool mask
  const float*         Adjd = (const float*)d_in[4];
  // d_in[5] = row_subtracted_distance_matrix : unused by the reference
  const float*         wAd  = (const float*)d_in[6];

  float* outd  = (float*)d_out;                              // (B,H,N,D)
  float* attnd = outd + (size_t)Bb * Hh * Nn * Dd;           // (B,H,N,N)

  dim3 grid(Bb * Hh * (Nn / 16));  // 4096 blocks
  dim3 block(256);                 // 8 wave32 waves
  hipLaunchKernelGGL(msrsa_attn_kernel, grid, block, 0, stream,
                     Qd, Kd, Vd, Md, Adjd, wAd, outd, attnd);
}
